// VIT_47029891891916
// MI455X (gfx1250) — matrix operand order repaired and checked by reading, not compiled, not run on hardware
//
#include <hip/hip_runtime.h>
#include <math.h>

// ---------------- model constants ----------------
#define DEPTH   12
#define DIM     768
#define NH      12
#define HD      64
#define IM      384
#define PATCHSZ 16
#define PGRID   24                 // 384/16 patches per side
#define NPATCH  576
#define NTOK    577
#define NCLS    1000
#define HID     3072
#define BATCH   8
#define MTOK    (BATCH * NTOK)     // 4616 rows (tokens)
#define MPAD    4624               // 4616 padded to 16
#define MPATCH  (BATCH * NPATCH)   // 4608 (multiple of 128)
#define NKPAD   608                // keys padded to 19*32
#define NBH     (BATCH * NH)       // 96 (batch,head) pairs

// ---------------- WMMA types (CDNA5, wave32) ----------------
typedef __attribute__((ext_vector_type(16))) __bf16 v16bf;
typedef __attribute__((ext_vector_type(8)))  __bf16 v8bf;
typedef __attribute__((ext_vector_type(8)))  float  v8f;

union V16 { v16bf v; v8bf h[2]; };

__device__ __forceinline__ unsigned short f2bf(float f) {
  unsigned int u = __float_as_uint(f);
  u += 0x7FFFu + ((u >> 16) & 1u);          // round-to-nearest-even
  return (unsigned short)(u >> 16);
}

__device__ __forceinline__ v8f wmma_bf16(v16bf a, v16bf b, v8f c) {
  // D(f32 16x16) = A(16x32 bf16) * B(32x16 bf16) + C
  return __builtin_amdgcn_wmma_f32_16x16x32_bf16(false, a, false, b, (short)0, c,
                                                 false, false);
}

// ---- CDNA5 async global->LDS copy (ASYNCcnt-tracked DMA, ISA §10 opcode 98) ----
__device__ __forceinline__ void async_ld_b128(unsigned lds_off, const void* gaddr) {
  asm volatile("global_load_async_to_lds_b128 %0, %1, off"
               :: "v"(lds_off), "v"(gaddr)
               : "memory");
}
__device__ __forceinline__ void wait_async0() {
  asm volatile("s_wait_asynccnt 0x0" ::: "memory");
}
// generic shared ptr -> 32-bit LDS byte offset (keeps the low 32 address bits)
__device__ __forceinline__ unsigned lds_off_of(const void* p) {
  return (unsigned)(unsigned long long)p;
}

// =====================================================================
// Generic bf16 GEMM:  C[M x Nvalid](f32) = A[M x K](bf16) * W[K x N](bf16) + bias
// W is provided TRANSPOSED: Wt[Npad x K] row-major (row = output col n).
// Block tile 128x64, BK=32, 8 waves (each wave -> 32x32 via 2x2 WMMA tiles).
// Both LDS tiles are staged with global_load_async_to_lds_b128.
// =====================================================================
#define BM 128
#define BN 64
#define BK 32
#define APITCH 48   // halves; 96B row pitch keeps 16B fragment slices aligned

__global__ __launch_bounds__(256)
void gemm_bf16_kernel(const unsigned short* __restrict__ A, int lda,
                      const unsigned short* __restrict__ Wt,   // [Npad][K]
                      const float* __restrict__ bias,
                      float* __restrict__ C, int ldc,
                      int M, int Nvalid, int K)
{
  __shared__ __align__(16) unsigned short As[BM * APITCH];
  __shared__ __align__(16) unsigned short Bs[BN * APITCH];

  const int tid  = threadIdx.x;
  const int lane = tid & 31;
  const int wave = tid >> 5;
  const int wm   = wave & 3;       // 4 waves along M
  const int wn   = wave >> 2;      // 2 waves along N
  const int half = lane >> 4;
  const int l16  = lane & 15;
  const int m0   = blockIdx.y * BM;
  const int n0   = blockIdx.x * BN;

  v8f acc[2][2] = {};

  for (int k0 = 0; k0 < K; k0 += BK) {
    // ---- stage A tile (128 rows x 32 halves): async DMA, 2x b128 per thread
    {
      const int r  = tid >> 1;                 // 0..127
      const int c0 = (tid & 1) * 16;           // 0 or 16 (halves)
      const int gr = m0 + r;
      const unsigned dst = lds_off_of(&As[r * APITCH + c0]);
      if (gr < M) {
        const unsigned short* src = A + (size_t)gr * lda + k0 + c0;
        async_ld_b128(dst,      src);
        async_ld_b128(dst + 16, src + 8);
      } else {
        v8bf z = {};
        *reinterpret_cast<v8bf*>(&As[r * APITCH + c0])     = z;
        *reinterpret_cast<v8bf*>(&As[r * APITCH + c0 + 8]) = z;
      }
    }
    // ---- stage Wt tile (64 rows x 32 halves): async DMA, 1x b128 per thread
    {
      const int n   = tid >> 2;                // 0..63
      const int seg = (tid & 3) * 8;           // 0,8,16,24 (halves)
      async_ld_b128(lds_off_of(&Bs[n * APITCH + seg]),
                    Wt + (size_t)(n0 + n) * K + k0 + seg);
    }
    wait_async0();
    __syncthreads();

    // ---- fragments ----
    V16 a[2], b[2];
    #pragma unroll
    for (int i = 0; i < 2; ++i) {
      const int r = wm * 32 + i * 16 + l16;      // A: lane row = M
      a[i].h[0] = *reinterpret_cast<const v8bf*>(&As[r * APITCH + 8 * half]);
      a[i].h[1] = *reinterpret_cast<const v8bf*>(&As[r * APITCH + 16 + 8 * half]);
    }
    #pragma unroll
    for (int j = 0; j < 2; ++j) {
      const int n = wn * 32 + j * 16 + l16;      // B: lane col = N, elems = K
      b[j].h[0] = *reinterpret_cast<const v8bf*>(&Bs[n * APITCH + 8 * half]);
      b[j].h[1] = *reinterpret_cast<const v8bf*>(&Bs[n * APITCH + 16 + 8 * half]);
    }
    #pragma unroll
    for (int i = 0; i < 2; ++i)
      #pragma unroll
      for (int j = 0; j < 2; ++j)
        acc[i][j] = wmma_bf16(a[i].v, b[j].v, acc[i][j]);
    __syncthreads();
  }

  // ---- epilogue: C layout, elem r -> M = r + 8*half, N = l16 ----
  #pragma unroll
  for (int i = 0; i < 2; ++i) {
    #pragma unroll
    for (int j = 0; j < 2; ++j) {
      const int col = n0 + wn * 32 + j * 16 + l16;
      if (col >= Nvalid) continue;
      const float bv = bias ? bias[col] : 0.0f;
      #pragma unroll
      for (int r = 0; r < 8; ++r) {
        const int row = m0 + wm * 32 + i * 16 + r + 8 * half;
        if (row < M) C[(size_t)row * ldc + col] = acc[i][j][r] + bv;
      }
    }
  }
}

// =====================================================================
// Fused attention: one wave per (b,h, 16-query tile).
// S = Q*K^T*scale (WMMA) -> exact softmax over 577 keys (row in LDS)
// -> O = P*V (WMMA).  Q/K packed [bh][608][64], V packed [bh][64][608].
// =====================================================================
__global__ __launch_bounds__(32)
void attention_kernel(const unsigned short* __restrict__ Qp,
                      const unsigned short* __restrict__ Kp,
                      const unsigned short* __restrict__ Vp,
                      float* __restrict__ O)
{
  __shared__ __align__(16) float          Srow[16 * NKPAD];   // 38.0 KB
  __shared__ __align__(16) unsigned short Pbuf[16 * NKPAD];   // 19.0 KB

  const int bh   = blockIdx.y;
  const int q0   = blockIdx.x * 16;
  const int lane = threadIdx.x;
  const int half = lane >> 4;
  const int l16  = lane & 15;

  const unsigned short* Qb = Qp + (size_t)bh * NKPAD * HD;
  const unsigned short* Kb = Kp + (size_t)bh * NKPAD * HD;
  const unsigned short* Vb = Vp + (size_t)bh * HD * NKPAD;

  // Q fragments (A-matrix): row = q0+l16, two 32-wide d chunks
  V16 aq[2];
  {
    const unsigned short* qr = Qb + (size_t)(q0 + l16) * HD;
    #pragma unroll
    for (int c = 0; c < 2; ++c) {
      aq[c].h[0] = *reinterpret_cast<const v8bf*>(&qr[32 * c + 8 * half]);
      aq[c].h[1] = *reinterpret_cast<const v8bf*>(&qr[32 * c + 16 + 8 * half]);
    }
  }

  const float scale = 0.125f;  // 1/sqrt(64)

  // ---- scores ----
  for (int kt = 0; kt < NKPAD / 16; ++kt) {
    V16 bk[2];
    const unsigned short* kr = Kb + (size_t)(kt * 16 + l16) * HD;
    #pragma unroll
    for (int c = 0; c < 2; ++c) {   // B: lane col = key, elems = d
      bk[c].h[0] = *reinterpret_cast<const v8bf*>(&kr[32 * c + 8 * half]);
      bk[c].h[1] = *reinterpret_cast<const v8bf*>(&kr[32 * c + 16 + 8 * half]);
    }
    v8f s = {};
    s = wmma_bf16(aq[0].v, bk[0].v, s);
    s = wmma_bf16(aq[1].v, bk[1].v, s);
    #pragma unroll
    for (int r = 0; r < 8; ++r)
      Srow[(r + 8 * half) * NKPAD + kt * 16 + l16] = s[r] * scale;
  }
  __syncthreads();

  // ---- softmax: 2 lanes per row (segments), combine with shfl_xor(16) ----
  {
    const int row = l16;
    const int beg = half * 304;
    const int end = (beg + 304 < NTOK) ? beg + 304 : NTOK;
    float mx = -1e30f;
    for (int k = beg; k < end; ++k) mx = fmaxf(mx, Srow[row * NKPAD + k]);
    mx = fmaxf(mx, __shfl_xor(mx, 16));
    float sum = 0.0f;
    for (int k = beg; k < end; ++k) {
      const float e = __expf(Srow[row * NKPAD + k] - mx);
      Srow[row * NKPAD + k] = e;
      sum += e;
    }
    sum += __shfl_xor(sum, 16);
    const float inv = 1.0f / sum;
    for (int k = beg; k < end; ++k)
      Pbuf[row * NKPAD + k] = f2bf(Srow[row * NKPAD + k] * inv);
    if (half == 1)  // zero padded keys 577..607
      for (int k = NTOK; k < NKPAD; ++k) Pbuf[row * NKPAD + k] = 0;
  }
  __syncthreads();

  // ---- O = P * V : 19 key-chunks of 32, 4 N-tiles covering hd=64 ----
  v8f o[4] = {};
  for (int c = 0; c < NKPAD / 32; ++c) {
    V16 ap;   // A: row = query l16, elems = keys
    ap.h[0] = *reinterpret_cast<const v8bf*>(&Pbuf[l16 * NKPAD + 32 * c + 8 * half]);
    ap.h[1] = *reinterpret_cast<const v8bf*>(&Pbuf[l16 * NKPAD + 32 * c + 16 + 8 * half]);
    #pragma unroll
    for (int j = 0; j < 4; ++j) {
      V16 bv;   // B: lane col = d, elems = keys (V stored d-major)
      const unsigned short* vr = Vb + (size_t)(j * 16 + l16) * NKPAD + 32 * c;
      bv.h[0] = *reinterpret_cast<const v8bf*>(&vr[8 * half]);
      bv.h[1] = *reinterpret_cast<const v8bf*>(&vr[16 + 8 * half]);
      o[j] = wmma_bf16(ap.v, bv.v, o[j]);
    }
  }

  // ---- store O into [B*577][768] at col h*64 ----
  const int b  = bh / NH;
  const int hh = bh % NH;
  #pragma unroll
  for (int j = 0; j < 4; ++j) {
    const int col = hh * HD + j * 16 + l16;
    #pragma unroll
    for (int r = 0; r < 8; ++r) {
      const int q = q0 + r + 8 * half;
      if (q < NTOK)
        O[((size_t)(b * NTOK + q)) * DIM + col] = o[j][r];
    }
  }
}

// =====================================================================
// Elementwise / data-movement kernels
// =====================================================================

// im2col patches -> bf16 [4608 x 768]
__global__ void patches_kernel(const float* __restrict__ x, unsigned short* __restrict__ out)
{
  const int idx = blockIdx.x * 256 + threadIdx.x;
  const int total = MPATCH * DIM;
  if (idx >= total) return;
  const int row = idx / DIM, col = idx % DIM;
  const int b = row / NPATCH, p = row % NPATCH;
  const int pr = p / PGRID, pc = p % PGRID;
  const int ch = col / 256, rem = col % 256;
  const int i = rem / PATCHSZ, j = rem % PATCHSZ;
  const float v = x[(((size_t)(b * 3 + ch)) * IM + pr * PATCHSZ + i) * IM + pc * PATCHSZ + j];
  out[idx] = f2bf(v);
}

// f32 weight -> bf16 TRANSPOSED tile: dst[n][k] over [Npad][K], zero-pad n>=N.
// transposed=0: src is [K][N] (normal layer weight); dst[n*K+k]=src[k*N+n]
// transposed=1: src is [N][K] (patch_w layout);      dst[n*K+k]=src[n*K+k]
__global__ void convert_wT_kernel(const float* __restrict__ src, unsigned short* __restrict__ dst,
                                  int K, int N, int Npad, int transposed)
{
  const int idx = blockIdx.x * 256 + threadIdx.x;
  if (idx >= Npad * K) return;
  const int n = idx / K, k = idx % K;
  float v = 0.0f;
  if (n < N) v = transposed ? src[(size_t)n * K + k] : src[(size_t)k * N + n];
  dst[idx] = f2bf(v);
}

// assemble h[b,t,:] = (t==0 ? cls : patch_out[b, t-1]) + pos_emb[t]
__global__ void assemble_h_kernel(const float* __restrict__ hp,
                                  const float* __restrict__ cls,
                                  const float* __restrict__ pos,
                                  float* __restrict__ h)
{
  const int idx = blockIdx.x * 256 + threadIdx.x;
  const int total = BATCH * NTOK * DIM;
  if (idx >= total) return;
  const int d = idx % DIM;
  const int t = (idx / DIM) % NTOK;
  const int b = idx / (DIM * NTOK);
  float v;
  if (t == 0) v = cls[d];
  else        v = hp[((size_t)(b * NPATCH + t - 1)) * DIM + d];
  h[idx] = v + pos[(size_t)t * DIM + d];
}

// per-row LayerNorm -> bf16; pads rows >= rowsValid with zeros.
__global__ __launch_bounds__(256)
void layernorm_bf16_kernel(const float* __restrict__ src, long srcRowStride, int rowsValid,
                           const float* __restrict__ g, const float* __restrict__ b,
                           float eps, unsigned short* __restrict__ out)
{
  __shared__ float red[256];
  const int row = blockIdx.x, tid = threadIdx.x;
  if (row >= rowsValid) {
    for (int c = tid; c < DIM; c += 256) out[(size_t)row * DIM + c] = 0;
    return;
  }
  const float* x = src + (size_t)row * srcRowStride;
  float s = 0.0f;
  for (int c = tid; c < DIM; c += 256) s += x[c];
  red[tid] = s; __syncthreads();
  for (int o = 128; o > 0; o >>= 1) { if (tid < o) red[tid] += red[tid + o]; __syncthreads(); }
  const float mean = red[0] * (1.0f / DIM);
  __syncthreads();
  float s2 = 0.0f;
  for (int c = tid; c < DIM; c += 256) { const float d = x[c] - mean; s2 += d * d; }
  red[tid] = s2; __syncthreads();
  for (int o = 128; o > 0; o >>= 1) { if (tid < o) red[tid] += red[tid + o]; __syncthreads(); }
  const float rstd = rsqrtf(red[0] * (1.0f / DIM) + eps);
  for (int c = tid; c < DIM; c += 256)
    out[(size_t)row * DIM + c] = f2bf((x[c] - mean) * rstd * g[c] + b[c]);
}

// split qkv (bias already added) into Q/K packed [bh][608][64] and V packed [bh][64][608]
__global__ void qkv_pack_kernel(const float* __restrict__ qkv,
                                unsigned short* __restrict__ Qp,
                                unsigned short* __restrict__ Kp,
                                unsigned short* __restrict__ Vp)
{
  const int idx = blockIdx.x * 256 + threadIdx.x;
  const int total = NBH * NKPAD * HD;
  if (idx >= total) return;
  const int d  = idx % HD;
  const int n  = (idx / HD) % NKPAD;
  const int bh = idx / (HD * NKPAD);
  const int b = bh / NH, hh = bh % NH;
  float qv = 0.0f, kv = 0.0f, vv = 0.0f;
  if (n < NTOK) {
    const float* base = qkv + ((size_t)(b * NTOK + n)) * (3 * DIM) + hh * HD + d;
    qv = base[0]; kv = base[DIM]; vv = base[2 * DIM];
  }
  Qp[idx] = f2bf(qv);
  Kp[idx] = f2bf(kv);
  Vp[((size_t)bh * HD + d) * NKPAD + n] = f2bf(vv);
}

// f32 [rows x cols] -> bf16, zero-padding rows >= rowsValid
__global__ void pack_rows_bf16_kernel(const float* __restrict__ src, unsigned short* __restrict__ dst,
                                      int rowsValid, int cols, int total)
{
  const int idx = blockIdx.x * 256 + threadIdx.x;
  if (idx >= total) return;
  const int row = idx / cols;
  dst[idx] = (row < rowsValid) ? f2bf(src[idx]) : (unsigned short)0;
}

// exact GELU then pack to bf16 (zero pad rows)
__global__ void gelu_pack_kernel(const float* __restrict__ src, unsigned short* __restrict__ dst,
                                 int rowsValid, int cols, int total)
{
  const int idx = blockIdx.x * 256 + threadIdx.x;
  if (idx >= total) return;
  const int row = idx / cols;
  if (row >= rowsValid) { dst[idx] = 0; return; }
  const float v = src[idx];
  dst[idx] = f2bf(0.5f * v * (1.0f + erff(v * 0.70710678118654752f)));
}

// h += c  (bias already folded into c by the GEMM)
__global__ void residual_add_kernel(float* __restrict__ h, const float* __restrict__ c, int total)
{
  const int idx = blockIdx.x * 256 + threadIdx.x;
  if (idx < total) h[idx] += c[idx];
}

// =====================================================================
// Host orchestration
// =====================================================================
static inline size_t align256(size_t x) { return (x + 255) & ~(size_t)255; }
static inline int cdiv(int a, int b) { return (a + b - 1) / b; }

extern "C" void kernel_launch(void* const* d_in, const int* in_sizes, int n_in,
                              void* d_out, int out_size, void* d_ws, size_t ws_size,
                              hipStream_t stream)
{
  (void)in_sizes; (void)n_in; (void)out_size; (void)ws_size;
  // ---- inputs (setup_inputs order) ----
  const float* x        = (const float*)d_in[0];
  const float* patch_w  = (const float*)d_in[1];
  const float* patch_b  = (const float*)d_in[2];
  const float* cls_tok  = (const float*)d_in[3];
  const float* pos_emb  = (const float*)d_in[4];
  const float* ln1_g    = (const float*)d_in[5];
  const float* ln1_b    = (const float*)d_in[6];
  const float* qkv_w    = (const float*)d_in[7];
  const float* qkv_b    = (const float*)d_in[8];
  const float* proj_w   = (const float*)d_in[9];
  const float* proj_b   = (const float*)d_in[10];
  const float* ln2_g    = (const float*)d_in[11];
  const float* ln2_b    = (const float*)d_in[12];
  const float* mlp_w1   = (const float*)d_in[13];
  const float* mlp_b1   = (const float*)d_in[14];
  const float* mlp_w2   = (const float*)d_in[15];
  const float* mlp_b2   = (const float*)d_in[16];
  const float* lnf_g    = (const float*)d_in[17];
  const float* lnf_b    = (const float*)d_in[18];
  const float* head_w   = (const float*)d_in[19];
  const float* head_b   = (const float*)d_in[20];
  float* out = (float*)d_out;

  // ---- workspace carve-up (~190 MB; aliased where lifetimes allow) ----
  char* ws = (char*)d_ws;
  size_t off = 0;
  auto carve = [&](size_t bytes) { char* p = ws + off; off += align256(bytes); return p; };

  float*          h_f32    = (float*)         carve((size_t)MPAD * DIM * 4);
  float*          tmp768   = (float*)         carve((size_t)MPAD * DIM * 4);   // patch out / proj out / mlp2 out
  unsigned short* y_bf     = (unsigned short*)carve((size_t)MPAD * DIM * 2);
  float*          qkv_f32  = (float*)         carve((size_t)MPAD * 3 * DIM * 4);
  unsigned short* Qp       = (unsigned short*)carve((size_t)NBH * NKPAD * HD * 2);
  unsigned short* Kp       = (unsigned short*)carve((size_t)NBH * NKPAD * HD * 2);
  unsigned short* Vp       = (unsigned short*)carve((size_t)NBH * HD * NKPAD * 2);
  unsigned short* o_bf     = (unsigned short*)carve((size_t)MPAD * DIM * 2);
  float*          hid_f32  = (float*)         carve((size_t)MPAD * HID * 4);
  unsigned short* hid_bf   = (unsigned short*)carve((size_t)MPAD * HID * 2);
  unsigned short* wbfT     = (unsigned short*)carve((size_t)DIM * HID * 2);    // largest weight tile (transposed)
  unsigned short* cls_bf   = (unsigned short*)carve((size_t)16 * DIM * 2);
  // aliases (lifetimes disjoint in stream order):
  float*          o_f32      = qkv_f32;   // attention out, after qkv_f32 consumed by pack
  unsigned short* patches_bf = hid_bf;    // patchify happens before any MLP

  const dim3 blk(256);
  auto g1 = [&](int total) { return dim3(cdiv(total, 256)); };

  // ================= patch embed =================
  patches_kernel<<<g1(MPATCH * DIM), blk, 0, stream>>>(x, patches_bf);
  convert_wT_kernel<<<g1(DIM * DIM), blk, 0, stream>>>(patch_w, wbfT, DIM, DIM, DIM, 1);
  gemm_bf16_kernel<<<dim3(DIM / BN, MPATCH / BM), blk, 0, stream>>>(
      patches_bf, DIM, wbfT, patch_b, tmp768, DIM, MPATCH, DIM, DIM);
  assemble_h_kernel<<<g1(BATCH * NTOK * DIM), blk, 0, stream>>>(tmp768, cls_tok, pos_emb, h_f32);

  // ================= transformer layers =================
  const int mTiles = cdiv(MTOK, BM);  // 37
  for (int l = 0; l < DEPTH; ++l) {
    const float* l1g = ln1_g + l * DIM;         const float* l1b = ln1_b + l * DIM;
    const float* qw  = qkv_w + (size_t)l * DIM * 3 * DIM;   const float* qb = qkv_b + l * 3 * DIM;
    const float* pw  = proj_w + (size_t)l * DIM * DIM;      const float* pb = proj_b + l * DIM;
    const float* l2g = ln2_g + l * DIM;         const float* l2b = ln2_b + l * DIM;
    const float* w1  = mlp_w1 + (size_t)l * DIM * HID;      const float* b1 = mlp_b1 + l * HID;
    const float* w2  = mlp_w2 + (size_t)l * HID * DIM;      const float* b2 = mlp_b2 + l * DIM;

    // --- attention half ---
    layernorm_bf16_kernel<<<dim3(MPAD), blk, 0, stream>>>(h_f32, DIM, MTOK, l1g, l1b, 1e-5f, y_bf);
    convert_wT_kernel<<<g1(3 * DIM * DIM), blk, 0, stream>>>(qw, wbfT, DIM, 3 * DIM, 3 * DIM, 0);
    gemm_bf16_kernel<<<dim3(3 * DIM / BN, mTiles), blk, 0, stream>>>(
        y_bf, DIM, wbfT, qb, qkv_f32, 3 * DIM, MTOK, 3 * DIM, DIM);
    qkv_pack_kernel<<<g1(NBH * NKPAD * HD), blk, 0, stream>>>(qkv_f32, Qp, Kp, Vp);
    attention_kernel<<<dim3(cdiv(NTOK, 16), NBH), dim3(32), 0, stream>>>(Qp, Kp, Vp, o_f32);
    pack_rows_bf16_kernel<<<g1(MPAD * DIM), blk, 0, stream>>>(o_f32, o_bf, MTOK, DIM, MPAD * DIM);
    convert_wT_kernel<<<g1(DIM * DIM), blk, 0, stream>>>(pw, wbfT, DIM, DIM, DIM, 0);
    gemm_bf16_kernel<<<dim3(DIM / BN, mTiles), blk, 0, stream>>>(
        o_bf, DIM, wbfT, pb, tmp768, DIM, MTOK, DIM, DIM);
    residual_add_kernel<<<g1(MTOK * DIM), blk, 0, stream>>>(h_f32, tmp768, MTOK * DIM);

    // --- MLP half ---
    layernorm_bf16_kernel<<<dim3(MPAD), blk, 0, stream>>>(h_f32, DIM, MTOK, l2g, l2b, 1e-5f, y_bf);
    convert_wT_kernel<<<g1(HID * DIM), blk, 0, stream>>>(w1, wbfT, DIM, HID, HID, 0);
    gemm_bf16_kernel<<<dim3(HID / BN, mTiles), blk, 0, stream>>>(
        y_bf, DIM, wbfT, b1, hid_f32, HID, MTOK, HID, DIM);
    gelu_pack_kernel<<<g1(MPAD * HID), blk, 0, stream>>>(hid_f32, hid_bf, MTOK, HID, MPAD * HID);
    convert_wT_kernel<<<g1(DIM * HID), blk, 0, stream>>>(w2, wbfT, HID, DIM, DIM, 0);
    gemm_bf16_kernel<<<dim3(DIM / BN, mTiles), blk, 0, stream>>>(
        hid_bf, HID, wbfT, b2, tmp768, DIM, MTOK, DIM, HID);
    residual_add_kernel<<<g1(MTOK * DIM), blk, 0, stream>>>(h_f32, tmp768, MTOK * DIM);
  }

  // ================= final LN on CLS rows + head =================
  // gather: row b reads h[b, 0, :] via row stride NTOK*DIM
  layernorm_bf16_kernel<<<dim3(16), blk, 0, stream>>>(h_f32, (long)NTOK * DIM, BATCH,
                                                      lnf_g, lnf_b, 1e-6f, cls_bf);
  convert_wT_kernel<<<g1(1024 * DIM), blk, 0, stream>>>(head_w, wbfT, DIM, NCLS, 1024, 0);
  gemm_bf16_kernel<<<dim3(1024 / BN, 1), blk, 0, stream>>>(
      cls_bf, DIM, wbfT, head_b, out, NCLS, BATCH, NCLS, DIM);
}
